// SemanticAlignmentModule_25366076850394
// MI455X (gfx1250) — matrix operand order repaired and checked by reading, not compiled, not run on hardware
//
#include <hip/hip_runtime.h>

#define DEVINL __device__ __forceinline__

typedef __attribute__((ext_vector_type(16))) __bf16       v16bf;
typedef __attribute__((ext_vector_type(8)))  float        v8f;
typedef __attribute__((ext_vector_type(4)))  unsigned int uivec4;

union FragAB { v16bf v; uivec4 q[2]; };

DEVINL unsigned short f32_to_bf16(float f) {
    unsigned int u = __float_as_uint(f);
    unsigned int r = u + 0x7FFFu + ((u >> 16) & 1u);
    return (unsigned short)(r >> 16);
}

DEVINL float wave_sum32(float s) {
#pragma unroll
    for (int off = 16; off > 0; off >>= 1) s += __shfl_xor(s, off, 32);
    return s;
}

// CDNA5 async copy: 16 bytes global -> LDS per lane, tracked by ASYNCcnt.
DEVINL void async_copy_b128(unsigned lds_off, const void* gptr) {
    unsigned long long g = (unsigned long long)(size_t)gptr;
    asm volatile("global_load_async_to_lds_b128 %0, %1, off"
                 :: "v"(lds_off), "v"(g) : "memory");
}
DEVINL void wait_async_le1() { asm volatile("s_wait_asynccnt 0x1" ::: "memory"); }
DEVINL void wait_async_le0() { asm volatile("s_wait_asynccnt 0x0" ::: "memory"); }

// ---------------------------------------------------------------------------
// LayerNorm1: visual (BN x 256) -> x_f32 (residual) and x_bf16 (GEMM operand)
// ---------------------------------------------------------------------------
__global__ __launch_bounds__(256)
void ln1_kernel(const float* __restrict__ vis, const float* __restrict__ g,
                const float* __restrict__ b, float* __restrict__ xf,
                unsigned short* __restrict__ xh) {
    const int row  = blockIdx.x * 8 + (threadIdx.x >> 5);
    const int lane = threadIdx.x & 31;
    const float* p = vis + (size_t)row * 256 + lane * 8;
    float v[8];
    float s = 0.f;
#pragma unroll
    for (int i = 0; i < 8; ++i) { v[i] = p[i]; s += v[i]; }
    s = wave_sum32(s);
    const float mean = s * (1.0f / 256.0f);
    float s2 = 0.f;
#pragma unroll
    for (int i = 0; i < 8; ++i) { float d = v[i] - mean; s2 += d * d; }
    s2 = wave_sum32(s2);
    const float rstd = rsqrtf(s2 * (1.0f / 256.0f) + 1e-5f);
    const int c0 = lane * 8;
#pragma unroll
    for (int i = 0; i < 8; ++i) {
        float y = (v[i] - mean) * rstd * g[c0 + i] + b[c0 + i];
        size_t idx = (size_t)row * 256 + c0 + i;
        xf[idx] = y;
        xh[idx] = f32_to_bf16(y);
    }
}

// ---------------------------------------------------------------------------
// Transpose + convert weight W (K x N, f32) -> Wt (N x K, bf16)
// ---------------------------------------------------------------------------
__global__ __launch_bounds__(256)
void wt_bf16_kernel(const float* __restrict__ W, unsigned short* __restrict__ Wt,
                    int K, int N) {
    long idx = (long)blockIdx.x * 256 + threadIdx.x;
    if (idx >= (long)K * N) return;
    int n = (int)(idx / K);
    int k = (int)(idx % K);
    Wt[idx] = f32_to_bf16(W[(size_t)k * N + n]);
}

// ---------------------------------------------------------------------------
// bf16 WMMA GEMM with async-LDS double-buffered B tile.
//   Block: 256 threads = 8 waves. Block tile 256(M) x 64(N).
//   Wave tile 32(M) x 64(N): 8 WMMAs per 32-deep K step.
//   B tile (64x32 bf16 = 4KB) staged via global_load_async_to_lds_b128,
//   ping-pong buffered; K loop processes tile PAIRS with two A register
//   sets (aX/aY) so the software pipeline needs no register copies.
//   All 4 B fragments are batch-loaded from LDS into distinct registers
//   before the 8-WMMA group so ds_load and WMMA issue overlap.
// MODE 0: outF = acc + bias                  (Q projection)
// MODE 1: outH = bf16(gelu_exact(acc+bias))  (FFN1)
// MODE 2: outF = resid + acc + bias          (FFN2 + residual)
// ---------------------------------------------------------------------------
DEVINL void load_a_frags(FragAB a[2], const unsigned short* r0,
                         const unsigned short* r1, int ko) {
    a[0].q[0] = *(const uivec4*)(r0 + ko);
    a[0].q[1] = *(const uivec4*)(r0 + ko + 8);
    a[1].q[0] = *(const uivec4*)(r1 + ko);
    a[1].q[1] = *(const uivec4*)(r1 + ko + 8);
}

DEVINL void wmma_step(const unsigned short* bb, int l16, int bkb,
                      const FragAB a[2], v8f acc[2][4]) {
    // Phase 1: batch all 8 ds_load_b128 into distinct registers
    FragAB bfr[4];
#pragma unroll
    for (int j = 0; j < 4; ++j) {
        const uivec4* bp = (const uivec4*)(bb + (j * 16 + l16) * 32 + bkb);
        bfr[j].q[0] = bp[0];
        bfr[j].q[1] = bp[1];
    }
    // Phase 2: 8 back-to-back WMMAs
#pragma unroll
    for (int j = 0; j < 4; ++j) {
        acc[0][j] = __builtin_amdgcn_wmma_f32_16x16x32_bf16(
            false, a[0].v, false, bfr[j].v, (short)0, acc[0][j], false, false);
        acc[1][j] = __builtin_amdgcn_wmma_f32_16x16x32_bf16(
            false, a[1].v, false, bfr[j].v, (short)0, acc[1][j], false, false);
    }
}

template <int MODE>
__global__ __launch_bounds__(256)
void gemm_bf16_wmma(const unsigned short* __restrict__ A,
                    const unsigned short* __restrict__ Bt,
                    const float* __restrict__ bias,
                    const float* __restrict__ resid,
                    float* __restrict__ outF,
                    unsigned short* __restrict__ outH,
                    int M, int N, int K) {
    __shared__ __align__(16) unsigned short bbuf[2][64 * 32];

    const int tid  = threadIdx.x;
    const int lane = tid & 31;
    const int wave = tid >> 5;
    const int n0   = blockIdx.x * 64;
    const int m0   = blockIdx.y * 256 + wave * 32;
    const int half = lane >> 4;      // 0: lanes 0-15, 1: lanes 16-31
    const int l16  = lane & 15;
    const int akb  = half * 16;      // A: K-half select per lane half
    const int bkb  = half * 16;      // B: K-half select per lane half

    // async B-tile copy mapping: each thread moves 16 contiguous bytes
    const int brow = tid >> 2;            // local n row 0..63
    const int bcol = (tid & 3) * 8;       // ushort offset within 32-wide K slice
    const unsigned short* bsrc = Bt + (size_t)(n0 + brow) * K + bcol;
    const unsigned lds_slot0 = (unsigned)(size_t)&bbuf[0][tid * 8];
    const unsigned lds_slot1 = (unsigned)(size_t)&bbuf[1][tid * 8];

    const unsigned short* arow0 = A + (size_t)(m0 + l16) * K;
    const unsigned short* arow1 = A + (size_t)(m0 + 16 + l16) * K;

    const int KT = K >> 5;   // even (8 or 32)

    // prologue: tile 0 in flight, A fragments for tile 0 in aX
    async_copy_b128(lds_slot0, bsrc);
    FragAB aX[2], aY[2];
    load_a_frags(aX, arow0, arow1, akb);

    v8f acc[2][4] = {};

    int kt = 0;
    for (; kt + 2 < KT; kt += 2) {
        // even tile kt (buf0); prefetch tile kt+1 -> buf1
        async_copy_b128(lds_slot1, bsrc + (size_t)(kt + 1) * 32);
        wait_async_le1();                 // tile kt landed (in-order)
        __syncthreads();
        load_a_frags(aY, arow0, arow1, (kt + 1) * 32 + akb);
        wmma_step(bbuf[0], l16, bkb, aX, acc);
        __syncthreads();                  // buf0 free for re-issue

        // odd tile kt+1 (buf1); prefetch tile kt+2 -> buf0
        async_copy_b128(lds_slot0, bsrc + (size_t)(kt + 2) * 32);
        wait_async_le1();                 // tile kt+1 landed
        __syncthreads();
        load_a_frags(aX, arow0, arow1, (kt + 2) * 32 + akb);
        wmma_step(bbuf[1], l16, bkb, aY, acc);
        __syncthreads();                  // buf1 free for re-issue
    }

    // epilogue pair: tiles KT-2 (buf0) and KT-1 (buf1)
    async_copy_b128(lds_slot1, bsrc + (size_t)(KT - 1) * 32);
    wait_async_le1();
    __syncthreads();
    load_a_frags(aY, arow0, arow1, (KT - 1) * 32 + akb);
    wmma_step(bbuf[0], l16, bkb, aX, acc);
    __syncthreads();
    wait_async_le0();
    __syncthreads();
    wmma_step(bbuf[1], l16, bkb, aY, acc);

#pragma unroll
    for (int mi = 0; mi < 2; ++mi) {
#pragma unroll
        for (int j = 0; j < 4; ++j) {
            const int col = n0 + j * 16 + l16;
            const float bs = bias[col];
#pragma unroll
            for (int r = 0; r < 8; ++r) {
                const int row = m0 + mi * 16 + half * 8 + r;
                float vacc = acc[mi][j][r] + bs;
                size_t idx = (size_t)row * N + col;
                if (MODE == 0) {
                    outF[idx] = vacc;
                } else if (MODE == 1) {
                    float gl = 0.5f * vacc *
                               (1.0f + erff(vacc * 0.7071067811865475f));
                    outH[idx] = f32_to_bf16(gl);
                } else {
                    outF[idx] = resid[idx] + vacc;
                }
            }
        }
    }
}

// ---------------------------------------------------------------------------
// K/V projection: text (616 x 512) @ k_w/v_w (512 x 256) + bias -> f32
// ---------------------------------------------------------------------------
__global__ __launch_bounds__(256)
void kv_proj_kernel(const float* __restrict__ text,
                    const float* __restrict__ kw, const float* __restrict__ kb,
                    const float* __restrict__ vw, const float* __restrict__ vb,
                    float* __restrict__ kf, float* __restrict__ vf) {
    const int row = blockIdx.x;
    const int n   = threadIdx.x;
    const float* t = text + (size_t)row * 512;
    float ak = 0.f, av = 0.f;
    for (int c = 0; c < 512; ++c) {
        float tv = t[c];
        ak = fmaf(tv, kw[(size_t)c * 256 + n], ak);
        av = fmaf(tv, vw[(size_t)c * 256 + n], av);
    }
    kf[(size_t)row * 256 + n] = ak + kb[n];
    vf[(size_t)row * 256 + n] = av + vb[n];
}

// pad mask: sum(|text_row|) <= 1e-6  -> 1.0 else 0.0 ; one wave per row
__global__ __launch_bounds__(256)
void pad_kernel(const float* __restrict__ text, float* __restrict__ pad, int rows) {
    const int row = blockIdx.x * 8 + (threadIdx.x >> 5);
    if (row >= rows) return;
    const int lane = threadIdx.x & 31;
    float s = 0.f;
    for (int c = lane; c < 512; c += 32) s += fabsf(text[(size_t)row * 512 + c]);
    s = wave_sum32(s);
    if (lane == 0) pad[row] = (s <= 1e-6f) ? 1.0f : 0.0f;
}

// in-place row L2 norm over 256 channels; one wave per row
__global__ __launch_bounds__(256)
void l2norm_kernel(float* __restrict__ x, int rows) {
    const int row = blockIdx.x * 8 + (threadIdx.x >> 5);
    if (row >= rows) return;
    const int lane = threadIdx.x & 31;
    float* p = x + (size_t)row * 256 + lane * 8;
    float v[8];
    float s = 0.f;
#pragma unroll
    for (int i = 0; i < 8; ++i) { v[i] = p[i]; s = fmaf(v[i], v[i], s); }
    s = wave_sum32(s);
    const float inv = 1.0f / fmaxf(sqrtf(s), 1e-6f);
#pragma unroll
    for (int i = 0; i < 8; ++i) p[i] = v[i] * inv;
}

// ---------------------------------------------------------------------------
// Top-5 sparse attention. One thread per (batch*token, head).
// ---------------------------------------------------------------------------
__global__ __launch_bounds__(256)
void attn_kernel(const float* __restrict__ q, const float* __restrict__ kf,
                 const float* __restrict__ vf, const float* __restrict__ pad,
                 const float* __restrict__ logit_scale,
                 float* __restrict__ aligned) {
    const int idx = blockIdx.x * 256 + threadIdx.x;   // BN*HEADS = 262144
    const int h   = idx & 7;
    const int bn  = idx >> 3;
    const int bb  = bn >> 12;                          // / 4096
    float ls = logit_scale[0];
    ls = fminf(fmaxf(ls, -2.0f), 2.0f);
    const float scale = expf(ls) * 0.17677669529663689f; // 1/sqrt(32)

    const float* qp = q + (size_t)bn * 256 + h * 32;
    float qv[32];
#pragma unroll
    for (int d = 0; d < 32; ++d) qv[d] = qp[d];

    const float* kp = kf + (size_t)bb * 77 * 256 + h * 32;
    const float* vp = vf + (size_t)bb * 77 * 256 + h * 32;
    const float* pp = pad + bb * 77;
    const float NEGINF = __int_as_float(0xff800000); // -inf

    float top[5];
#pragma unroll
    for (int i = 0; i < 5; ++i) top[i] = NEGINF;

    for (int t = 0; t < 77; ++t) {
        float sim;
        if (pp[t] != 0.f) {
            sim = NEGINF;
        } else {
            const float* kr = kp + (size_t)t * 256;
            float s = 0.f;
#pragma unroll
            for (int d = 0; d < 32; ++d) s = fmaf(qv[d], kr[d], s);
            sim = s * scale;
        }
        if (sim > top[4]) {
            top[4] = sim;
#pragma unroll
            for (int i = 4; i > 0; --i) {
                if (top[i] > top[i - 1]) {
                    float tmp = top[i - 1]; top[i - 1] = top[i]; top[i] = tmp;
                }
            }
        }
    }
    const float kth = top[4];
    const float mx  = top[0];

    float denom = 0.f;
    float acc[32];
#pragma unroll
    for (int d = 0; d < 32; ++d) acc[d] = 0.f;

    for (int t = 0; t < 77; ++t) {
        float sim;
        if (pp[t] != 0.f) {
            sim = NEGINF;
        } else {
            const float* kr = kp + (size_t)t * 256;
            float s = 0.f;
#pragma unroll
            for (int d = 0; d < 32; ++d) s = fmaf(qv[d], kr[d], s);
            sim = s * scale;
        }
        if (sim >= kth) {
            float w = expf(sim - mx);
            denom += w;
            const float* vr = vp + (size_t)t * 256;
#pragma unroll
            for (int d = 0; d < 32; ++d) acc[d] = fmaf(w, vr[d], acc[d]);
        }
    }
    const float inv = 1.0f / denom;
    float* op = aligned + (size_t)bn * 256 + h * 32;
#pragma unroll
    for (int d = 0; d < 32; ++d) op[d] = acc[d] * inv;
}

// ---------------------------------------------------------------------------
// gate + gated residual + LayerNorm2 ; one wave per row
// ---------------------------------------------------------------------------
__global__ __launch_bounds__(256)
void gate_ln2_kernel(const float* __restrict__ xf, const float* __restrict__ aligned,
                     const float* __restrict__ gw, const float* __restrict__ gb,
                     const float* __restrict__ alpha_p,
                     const float* __restrict__ g2, const float* __restrict__ b2,
                     float* __restrict__ yf, unsigned short* __restrict__ yh) {
    const int row  = blockIdx.x * 8 + (threadIdx.x >> 5);
    const int lane = threadIdx.x & 31;
    const int c0   = lane * 8;
    const float* xp = xf + (size_t)row * 256 + c0;
    const float* ap = aligned + (size_t)row * 256 + c0;
    float xv[8], av[8];
    float dot = 0.f;
#pragma unroll
    for (int i = 0; i < 8; ++i) {
        xv[i] = xp[i];
        av[i] = ap[i];
        dot = fmaf(xv[i], gw[c0 + i], dot);
    }
    dot = wave_sum32(dot);
    const float gate = 1.0f / (1.0f + expf(-(dot + gb[0])));
    const float coef = alpha_p[0] * gate;

    float yv[8];
    float s = 0.f;
#pragma unroll
    for (int i = 0; i < 8; ++i) { yv[i] = xv[i] + coef * av[i]; s += yv[i]; }
    s = wave_sum32(s);
    const float mean = s * (1.0f / 256.0f);
    float s2 = 0.f;
#pragma unroll
    for (int i = 0; i < 8; ++i) { float d = yv[i] - mean; s2 += d * d; }
    s2 = wave_sum32(s2);
    const float rstd = rsqrtf(s2 * (1.0f / 256.0f) + 1e-5f);
#pragma unroll
    for (int i = 0; i < 8; ++i) {
        float o = (yv[i] - mean) * rstd * g2[c0 + i] + b2[c0 + i];
        size_t idx = (size_t)row * 256 + c0 + i;
        yf[idx] = o;
        yh[idx] = f32_to_bf16(o);
    }
}

// ---------------------------------------------------------------------------
extern "C" void kernel_launch(void* const* d_in, const int* in_sizes, int n_in,
                              void* d_out, int out_size, void* d_ws, size_t ws_size,
                              hipStream_t stream) {
    const float* visual = (const float*)d_in[0];   // (8,64,64,256)
    const float* text   = (const float*)d_in[1];   // (8,77,512)
    const float* n1_g   = (const float*)d_in[2];
    const float* n1_b   = (const float*)d_in[3];
    const float* n2_g   = (const float*)d_in[4];
    const float* n2_b   = (const float*)d_in[5];
    const float* q_w    = (const float*)d_in[6];   // (256,256)
    const float* q_b    = (const float*)d_in[7];
    const float* k_w    = (const float*)d_in[8];   // (512,256)
    const float* k_b    = (const float*)d_in[9];
    const float* v_w    = (const float*)d_in[10];  // (512,256)
    const float* v_b    = (const float*)d_in[11];
    const float* gate_w = (const float*)d_in[12];  // (256,1)
    const float* gate_b = (const float*)d_in[13];
    const float* lscale = (const float*)d_in[14];
    const float* f1_w   = (const float*)d_in[15];  // (256,1024)
    const float* f1_b   = (const float*)d_in[16];
    const float* f2_w   = (const float*)d_in[17];  // (1024,256)
    const float* f2_b   = (const float*)d_in[18];
    const float* alpha  = (const float*)d_in[19];
    float* out          = (float*)d_out;

    const int BN = 32768;    // 8 * 64 * 64
    const int BT = 616;      // 8 * 77
    const int C  = 256;
    const int F  = 1024;

    size_t off = 0;
    auto carve = [&](size_t bytes) -> void* {
        void* p = (char*)d_ws + off;
        off += (bytes + 255) & ~(size_t)255;
        return p;
    };
    float*          x_f32  = (float*)carve((size_t)BN * C * 4);
    unsigned short* x_bf16 = (unsigned short*)carve((size_t)BN * C * 2);
    float*          q_full = (float*)carve((size_t)BN * C * 4);
    float*          k_full = (float*)carve((size_t)BT * C * 4);
    float*          v_full = (float*)carve((size_t)BT * C * 4);
    float*          padm   = (float*)carve((size_t)BT * 4);
    float*          align_ = (float*)carve((size_t)BN * C * 4);
    float*          y_f32  = (float*)carve((size_t)BN * C * 4);
    unsigned short* y_bf16 = (unsigned short*)carve((size_t)BN * C * 2);
    unsigned short* h_bf16 = (unsigned short*)carve((size_t)BN * F * 2);
    unsigned short* q_wt   = (unsigned short*)carve((size_t)C * C * 2);
    unsigned short* f1_wt  = (unsigned short*)carve((size_t)C * F * 2);
    unsigned short* f2_wt  = (unsigned short*)carve((size_t)F * C * 2);
    (void)ws_size; (void)n_in; (void)in_sizes; (void)out_size;

    // 1) LayerNorm1 -> x (f32 + bf16)
    ln1_kernel<<<BN / 8, 256, 0, stream>>>(visual, n1_g, n1_b, x_f32, x_bf16);

    // 2) weight conversions (f32 KxN -> bf16 NxK)
    wt_bf16_kernel<<<(C * C + 255) / 256, 256, 0, stream>>>(q_w, q_wt, C, C);
    wt_bf16_kernel<<<(C * F + 255) / 256, 256, 0, stream>>>(f1_w, f1_wt, C, F);
    wt_bf16_kernel<<<(F * C + 255) / 256, 256, 0, stream>>>(f2_w, f2_wt, F, C);

    // 3) K/V projections + pad mask (tiny, VALU)
    kv_proj_kernel<<<BT, 256, 0, stream>>>(text, k_w, k_b, v_w, v_b, k_full, v_full);
    pad_kernel<<<(BT + 7) / 8, 256, 0, stream>>>(text, padm, BT);

    // 4) Q projection: x (32768x256) @ q_w (256x256) -> q_full  [WMMA+async]
    gemm_bf16_wmma<0><<<dim3(C / 64, BN / 256), 256, 0, stream>>>(
        x_bf16, q_wt, q_b, nullptr, q_full, nullptr, BN, C, C);

    // 5) L2 norms (in place)
    l2norm_kernel<<<BN / 8, 256, 0, stream>>>(q_full, BN);
    l2norm_kernel<<<(BT + 7) / 8, 256, 0, stream>>>(k_full, BT);

    // 6) top-5 sparse attention -> aligned
    attn_kernel<<<(BN * 8) / 256, 256, 0, stream>>>(q_full, k_full, v_full,
                                                    padm, lscale, align_);

    // 7) gate + residual + LayerNorm2 -> y (f32 + bf16)
    gate_ln2_kernel<<<BN / 8, 256, 0, stream>>>(x_f32, align_, gate_w, gate_b,
                                                alpha, n2_g, n2_b, y_f32, y_bf16);

    // 8) FFN1: y (32768x256) @ f1_w (256x1024), exact GELU -> h  [WMMA+async]
    gemm_bf16_wmma<1><<<dim3(F / 64, BN / 256), 256, 0, stream>>>(
        y_bf16, f1_wt, f1_b, nullptr, nullptr, h_bf16, BN, F, C);

    // 9) FFN2: h (32768x1024) @ f2_w (1024x256) + y resid -> out [WMMA+async]
    gemm_bf16_wmma<2><<<dim3(C / 64, BN / 256), 256, 0, stream>>>(
        h_bf16, f2_wt, f2_b, y_f32, out, nullptr, BN, C, F);
}
